// MoEGate_1692217114679
// MI455X (gfx1250) — compile-verified, matrix operand order checked by reading, not run on hardware
//
#include <hip/hip_runtime.h>

#define B_BATCH 4
#define S_LEN   4096
#define D_DIM   2048
#define E_EXP   64
#define M_ROWS  (B_BATCH * S_LEN)   // 16384
#define NCHUNK  (D_DIM / 32)        // 64 K-chunks of 32
#define ALPHA   0.01f

typedef __attribute__((ext_vector_type(16))) __bf16 v16bf;
typedef __attribute__((ext_vector_type(4)))  __bf16 v4bf;
typedef __attribute__((ext_vector_type(8)))  float  v8f;
typedef __attribute__((ext_vector_type(4)))  float  v4f;

// ---------------------------------------------------------------------------
// Fragment-order helpers.  For the 16-bit A/B WMMA layout, lane l of a wave
// holds, for K-chunk c, the 16 values {k = c*32 + h*8 + g*16 + j ; h=l>>4,
// g=0..1, j=0..7} of row/col (l&15).  We store A (LDS) and B (global ws) as
// frag[c][lane][16] so hot-loop loads are contiguous 32B per lane:
// conflict-free in LDS, fully coalesced (1 KB/wave/step) in global.
// ---------------------------------------------------------------------------

// Kernel 0: build fragment-ordered bf16 weight. id = (tile t, chunk c, lane l)
__global__ __launch_bounds__(256)
void k_prepare_weight(const float* __restrict__ w, __bf16* __restrict__ wbt) {
    const int id = blockIdx.x * 256 + threadIdx.x;   // 0 .. 4*64*32-1
    const int l  = id & 31;
    const int c  = (id >> 5) & (NCHUNK - 1);
    const int t  = id >> 11;                          // expert tile 0..3
    const int n  = t * 16 + (l & 15);                 // expert (B column)
    const int h  = l >> 4;
    const float* wr = w + (size_t)n * D_DIM + c * 32 + h * 8;
    __bf16* dst = wbt + ((size_t)id << 4);
#pragma unroll
    for (int j = 0; j < 8; ++j) dst[j]     = (__bf16)wr[j];        // group 0
#pragma unroll
    for (int j = 0; j < 8; ++j) dst[j + 8] = (__bf16)wr[j + 16];   // group 1
}

// ---------------------------------------------------------------------------
// Kernel 1: logits = x @ W^T  (M=16384, K=2048, N=64) via WMMA bf16.
// Block = 128 threads (4 waves); block owns 16 rows, wave w owns experts
// [16w,16w+16).  x staged NT f32 -> bf16 into fragment-ordered LDS.
// ---------------------------------------------------------------------------
__global__ __launch_bounds__(128)
void k_gate_gemm(const float* __restrict__ x, const __bf16* __restrict__ wbt,
                 float* __restrict__ logits) {
    __shared__ __bf16 ldsA[16 * D_DIM];               // 64 KB, frag[64][32][16]

    const int tid  = threadIdx.x;
    const int row0 = blockIdx.x * 16;

    // ---- stage A tile: one NT streaming pass over x, scatter to frag order
    const v4f* src = reinterpret_cast<const v4f*>(x + (size_t)row0 * D_DIM);
    for (int i = tid; i < 16 * (D_DIM / 4); i += 128) {
        v4f v = __builtin_nontemporal_load(src + i);
        const int m  = i >> 9;            // row in tile
        const int k  = (i & 511) << 2;    // starting K of this float4
        const int c  = k >> 5;            // K-chunk
        const int kc = k & 31;
        const int h  = (kc >> 3) & 1;     // lane half
        const int j  = (kc & 7) + ((kc >> 4) << 3);   // position in fragment
        const int lane = (h << 4) + m;
        v4bf o;
        o[0] = (__bf16)v[0]; o[1] = (__bf16)v[1];
        o[2] = (__bf16)v[2]; o[3] = (__bf16)v[3];
        *reinterpret_cast<v4bf*>(ldsA + (((size_t)(c * 32 + lane)) << 4) + j) = o;
    }
    __syncthreads();

    const int wave = tid >> 5;
    const int lane = tid & 31;

    const v16bf* Af = reinterpret_cast<const v16bf*>(ldsA);
    const v16bf* Bf = reinterpret_cast<const v16bf*>(wbt) + (size_t)wave * NCHUNK * 32;

    v8f acc = {};
    for (int c = 0; c < NCHUNK; ++c) {
        v16bf a = Af[c * 32 + lane];      // contiguous 32B, bank-conflict-free
        v16bf b = Bf[c * 32 + lane];      // contiguous 32B, coalesced across wave
        acc = __builtin_amdgcn_wmma_f32_16x16x32_bf16(
            /*neg_a=*/false, a, /*neg_b=*/false, b,
            /*c_mod=*/(short)0, acc, /*reuse_a=*/false, /*reuse_b=*/false);
    }

    // 32-bit C/D 16x16 layout: VGPR v -> M = v + 8*(lane>>4), N = lane&15
    const int m    = lane & 15;
    const int half = lane >> 4;
#pragma unroll
    for (int v = 0; v < 8; ++v) {
        logits[(size_t)(row0 + half * 8 + v) * E_EXP + wave * 16 + m] = acc[v];
    }
}

// ---------------------------------------------------------------------------
// Kernel 2: softmax over 64 experts, top-2 + renorm, aux-stat accumulation.
// One wave (32 lanes) per token; lane holds experts {lane, lane+32}.
// ---------------------------------------------------------------------------
__global__ __launch_bounds__(256)
void k_softmax_topk(const float* __restrict__ logits,
                    int* __restrict__ idx_out, float* __restrict__ w_out,
                    float* __restrict__ msum, float* __restrict__ cnt) {
    __shared__ float lds_ms[E_EXP];
    __shared__ float lds_ct[E_EXP];

    const int tid = threadIdx.x;
    if (tid < E_EXP) { lds_ms[tid] = 0.0f; lds_ct[tid] = 0.0f; }
    __syncthreads();

    const int wave  = tid >> 5;
    const int lane  = tid & 31;
    const int token = blockIdx.x * 8 + wave;

    float l0 = logits[(size_t)token * E_EXP + lane];
    float l1 = logits[(size_t)token * E_EXP + lane + 32];

    // softmax (max-subtracted), wave32 butterfly reductions
    float mx = fmaxf(l0, l1);
#pragma unroll
    for (int off = 16; off; off >>= 1) mx = fmaxf(mx, __shfl_xor(mx, off, 32));
    float e0 = __expf(l0 - mx), e1 = __expf(l1 - mx);
    float sum = e0 + e1;
#pragma unroll
    for (int off = 16; off; off >>= 1) sum += __shfl_xor(sum, off, 32);
    float p0 = e0 / sum, p1 = e1 / sum;

    // top-1 (ties -> lowest index, matching lax.top_k)
    float bs; int bi;
    if (p0 >= p1) { bs = p0; bi = lane; } else { bs = p1; bi = lane + 32; }
#pragma unroll
    for (int off = 16; off; off >>= 1) {
        float ov = __shfl_xor(bs, off, 32);
        int   oi = __shfl_xor(bi, off, 32);
        if (ov > bs || (ov == bs && oi < bi)) { bs = ov; bi = oi; }
    }
    // top-2: exclude winner, reduce again
    float ss = -1.0f; int si = E_EXP;
    if (bi != lane) { ss = p0; si = lane; }
    if (bi != lane + 32 && (p1 > ss || (p1 == ss && lane + 32 < si))) { ss = p1; si = lane + 32; }
#pragma unroll
    for (int off = 16; off; off >>= 1) {
        float ov = __shfl_xor(ss, off, 32);
        int   oi = __shfl_xor(si, off, 32);
        if (ov > ss || (ov == ss && oi < si)) { ss = ov; si = oi; }
    }

    // aux statistics: block-local LDS accumulation (ds_add_f32)
    atomicAdd(&lds_ms[lane],      p0);
    atomicAdd(&lds_ms[lane + 32], p1);
    if (lane == 0) {
        float denom = bs + ss + 1e-20f;
        idx_out[token * 2 + 0] = bi;
        idx_out[token * 2 + 1] = si;
        w_out[token * 2 + 0] = bs / denom;
        w_out[token * 2 + 1] = ss / denom;
        atomicAdd(&lds_ct[bi], 1.0f);
        atomicAdd(&lds_ct[si], 1.0f);
    }
    __syncthreads();

    // 8 tokens per block, 4096 tokens per batch -> block is batch-uniform
    const int b = (blockIdx.x * 8) / S_LEN;
    if (tid < E_EXP) {
        atomicAdd(&msum[b * E_EXP + tid], lds_ms[tid]);
        atomicAdd(&cnt [b * E_EXP + tid], lds_ct[tid]);
    }
}

// ---------------------------------------------------------------------------
// Kernel 3: aux_loss = ALPHA * mean_b( sum_e( ce[b,e] * mean_scores[b,e] ) )
// ce = cnt * E/(S*K);  mean_scores = msum / S.  256 entries -> 1 block.
// ---------------------------------------------------------------------------
__global__ __launch_bounds__(256)
void k_aux(const float* __restrict__ msum, const float* __restrict__ cnt,
           float* __restrict__ out_aux) {
    __shared__ float red[256];
    const int t = threadIdx.x;                      // t = b*64 + e
    const float ce = cnt[t] * ((float)E_EXP / (float)(S_LEN * 2));
    const float ms = msum[t] / (float)S_LEN;
    red[t] = ce * ms;
    __syncthreads();
#pragma unroll
    for (int s = 128; s; s >>= 1) {
        if (t < s) red[t] += red[t + s];
        __syncthreads();
    }
    if (t == 0) out_aux[0] = red[0] * (ALPHA / (float)B_BATCH);
}

// ---------------------------------------------------------------------------
extern "C" void kernel_launch(void* const* d_in, const int* in_sizes, int n_in,
                              void* d_out, int out_size, void* d_ws, size_t ws_size,
                              hipStream_t stream) {
    const float* x = (const float*)d_in[0];   // [4,4096,2048] f32
    const float* w = (const float*)d_in[1];   // [64,2048]     f32

    char* ws = (char*)d_ws;
    __bf16* wbt    = (__bf16*)ws;                                   // 256 KB, frag order
    float*  logits = (float*)(ws + 262144);                         // 4 MB
    float*  msum   = (float*)(ws + 262144 + (size_t)M_ROWS * E_EXP * 4);
    float*  cnt    = msum + B_BATCH * E_EXP;

    int*   idx_out = (int*)d_out;                     // [16384,2] as int32 bits
    float* w_out   = (float*)d_out + M_ROWS * 2;      // [16384,2]
    float* aux     = (float*)d_out + M_ROWS * 4;      // scalar

    hipMemsetAsync(msum, 0, 2 * B_BATCH * E_EXP * sizeof(float), stream);

    k_prepare_weight<<<(4 * NCHUNK * 32) / 256, 256, 0, stream>>>(w, wbt);
    k_gate_gemm<<<M_ROWS / 16, 128, 0, stream>>>(x, wbt, logits);
    k_softmax_topk<<<M_ROWS / 8, 256, 0, stream>>>(logits, idx_out, w_out, msum, cnt);
    k_aux<<<1, 256, 0, stream>>>(msum, cnt, aux);
}
